// MyRNN_28260884807880
// MI455X (gfx1250) — matrix operand order repaired and checked by reading, not compiled, not run on hardware
//
#include <hip/hip_runtime.h>

// Problem constants (match reference)
#define B_    4096
#define T_    80
#define V_    10000
#define E_    100
#define U_    1024
#define EP_   128      // E padded to multiple of 32 for WMMA K
#define ROWS_ 32       // batch rows per block
#define NT_   16       // 16x16 col tiles per wave (16*16 = 256 cols)

typedef __attribute__((ext_vector_type(16))) __bf16 v16bf;
typedef __attribute__((ext_vector_type(8)))  __bf16 v8bf;
typedef __attribute__((ext_vector_type(8)))  float  v8f;
typedef __attribute__((ext_vector_type(4)))  int    v4i;

// gfx1250 hardware tanh (TRANS op, co-executes with matrix pipe) if available
#if __has_builtin(__builtin_amdgcn_tanhf)
#define TANHF_(x) __builtin_amdgcn_tanhf(x)
#else
#define TANHF_(x) tanhf(x)
#endif

#if __has_builtin(__builtin_amdgcn_global_load_async_to_lds_b128)
#define HAVE_ASYNC_LDS_ 1
typedef __attribute__((address_space(1))) v4i* gv4i_p;   // global (AS1)
typedef __attribute__((address_space(3))) v4i* lv4i_p;   // LDS (AS3)
#endif

// ---------- prep: convert/transpose weights to padded bf16 in workspace ----------

__global__ void prep_emb_k(const float* __restrict__ emb, __bf16* __restrict__ embP) {
    int i = blockIdx.x * blockDim.x + threadIdx.x;
    if (i >= V_ * EP_) return;
    int r = i >> 7, c = i & (EP_ - 1);
    embP[i] = (c < E_) ? (__bf16)emb[r * E_ + c] : (__bf16)0.0f;
}

// WxT[u][e] = Wx[e][u], zero-padded e in [100,128)
__global__ void prep_wxt_k(const float* __restrict__ Wx, __bf16* __restrict__ WxT) {
    int i = blockIdx.x * blockDim.x + threadIdx.x;
    if (i >= U_ * EP_) return;
    int u = i >> 7, e = i & (EP_ - 1);
    WxT[i] = (e < E_) ? (__bf16)Wx[e * U_ + u] : (__bf16)0.0f;
}

// WhT[n][k] = Wh[k][n]
__global__ void prep_wht_k(const float* __restrict__ Wh, __bf16* __restrict__ WhT) {
    int i = blockIdx.x * blockDim.x + threadIdx.x;
    if (i >= U_ * U_) return;
    int n = i >> 10, k = i & (U_ - 1);
    WhT[i] = (__bf16)Wh[k * U_ + n];
}

// ---------- main persistent RNN kernel ----------
// grid = B/ROWS blocks, 256 threads = 8 wave32 (2 row-waves x 4 col-waves).
// LDS: h tile [32][1024] bf16 (64KB) + x_t tile [32][128] bf16 (8KB) = 72KB dynamic.

__global__ __launch_bounds__(256) void rnn_main_k(
    const int*   __restrict__ inputs,   // [B,T]
    const __bf16* __restrict__ embP,    // [V][128]
    const __bf16* __restrict__ WxT,     // [1024][128]  (N-major)
    const __bf16* __restrict__ WhT,     // [1024][1024] (N-major)
    const float* __restrict__ bg,       // [1024]
    const float* __restrict__ Wo,       // [1024]
    const float* __restrict__ bo,       // [1]
    float*       __restrict__ out)      // [B]
{
    extern __shared__ __align__(16) char smem[];
    __bf16 (*hbuf)[U_]  = (__bf16(*)[U_])smem;
    __bf16 (*xs)[EP_]   = (__bf16(*)[EP_])(smem + (size_t)ROWS_ * U_ * 2);

    const int tid   = threadIdx.x;
    const int lane  = tid & 31;
    const int wave  = tid >> 5;
    const int wr    = wave >> 2;            // 0..1 : row-wave
    const int wc    = wave & 3;             // 0..3 : col-wave
    const int r0    = wr * 16;              // row base within block tile
    const int c0    = wc * 256;             // col base within U
    const int lhalf = lane >> 4;            // 0/1
    const int lmod  = lane & 15;
    const int rowbase = blockIdx.x * ROWS_;

    // per-lane bias for the 16 col tiles (bias is indexed by output col N)
    float bias[NT_];
    #pragma unroll
    for (int ct = 0; ct < NT_; ++ct) bias[ct] = bg[c0 + ct * 16 + lmod];

    // h0 = 0
    {
        v8bf z = {};
        for (int i = tid; i < ROWS_ * U_ / 8; i += 256) ((v8bf*)smem)[i] = z;
    }
    __syncthreads();

    const int grow = rowbase + (tid >> 3);  // batch row this thread gathers
    const int seg  = tid & 7;               // 16-element segment of the 128 cols

    // warm the first timestep's embedding rows (global_prefetch_b8)
    {
        int idx0 = inputs[grow * T_];
        __builtin_prefetch(embP + (size_t)idx0 * EP_ + seg * 16, 0, 1);
    }

    for (int t = 0; t < T_; ++t) {
        // ---- gather x_t = embP[inputs[:,t]] into LDS (bf16, padded) ----
        {
            int idx = inputs[grow * T_ + t];
            const __bf16* srcp = embP + (size_t)idx * EP_ + seg * 16;
            __bf16* dstp = &xs[tid >> 3][seg * 16];
#ifdef HAVE_ASYNC_LDS_
            // CDNA5 async global->LDS copy (bypasses VGPRs, tracked by ASYNCcnt).
            // inttoptr casts: AS1 == numeric global addr; AS3 == low 32 bits of
            // the generic LDS pointer (same as LLVM's addrspacecast).
            gv4i_p g0 = (gv4i_p)(unsigned long long)(size_t)srcp;
            gv4i_p g1 = (gv4i_p)(unsigned long long)(size_t)(srcp + 8);
            lv4i_p l0 = (lv4i_p)(unsigned)(size_t)dstp;
            lv4i_p l1 = (lv4i_p)(unsigned)(size_t)(dstp + 8);
            __builtin_amdgcn_global_load_async_to_lds_b128(g0, l0, 0, 0);
            __builtin_amdgcn_global_load_async_to_lds_b128(g1, l1, 0, 0);
#else
            ((v8bf*)dstp)[0] = ((const v8bf*)srcp)[0];
            ((v8bf*)dstp)[1] = ((const v8bf*)srcp)[1];
#endif
            // prefetch next timestep's embedding rows into cache
            if (t + 1 < T_) {
                int idxn = inputs[grow * T_ + t + 1];
                __builtin_prefetch(embP + (size_t)idxn * EP_ + seg * 16, 0, 1);
            }
#ifdef HAVE_ASYNC_LDS_
#if __has_builtin(__builtin_amdgcn_s_wait_asynccnt)
            __builtin_amdgcn_s_wait_asynccnt(0);
#else
            asm volatile("s_wait_asynccnt 0" ::: "memory");
#endif
#endif
        }
        __syncthreads();   // xs ready; also: prev step's h writes visible

        const v8f zf = {};
        v8f acc[NT_];
        #pragma unroll
        for (int ct = 0; ct < NT_; ++ct) acc[ct] = zf;

        // ---- acc += x_t @ WxT  (K = 0..127) ----
        #pragma unroll 1
        for (int kc = 0; kc < EP_ / 32; ++kc) {
            const int k0 = kc * 32;
            v16bf a;   // A 16x32 bf16: lanes0-15 K {0..7,16..23}, lanes16-31 K {8..15,24..31}
            const __bf16* ap = &xs[r0 + lmod][k0 + 8 * lhalf];
            ((v8bf*)&a)[0] = *(const v8bf*)(ap);
            ((v8bf*)&a)[1] = *(const v8bf*)(ap + 16);
            const int kb = k0 + 8 * lhalf;
            #pragma unroll
            for (int ct = 0; ct < NT_; ++ct) {
                const __bf16* bp = WxT + (size_t)(c0 + ct * 16 + lmod) * EP_ + kb;
                v16bf bm;  // B 32x16 bf16: lane holds col N, K {0..7,16..23} (+8 for lanes16-31)
                ((v8bf*)&bm)[0] = *(const v8bf*)(bp);
                ((v8bf*)&bm)[1] = *(const v8bf*)(bp + 16);
                acc[ct] = __builtin_amdgcn_wmma_f32_16x16x32_bf16(
                    false, a, false, bm, (short)0, acc[ct], false, false);
            }
        }

        // ---- acc += h @ WhT  (K = 0..1023) ----
        #pragma unroll 1
        for (int kc = 0; kc < U_ / 32; ++kc) {
            const int k0 = kc * 32;
            v16bf a;
            const __bf16* ap = &hbuf[r0 + lmod][k0 + 8 * lhalf];
            ((v8bf*)&a)[0] = *(const v8bf*)(ap);
            ((v8bf*)&a)[1] = *(const v8bf*)(ap + 16);
            const int kb = k0 + 8 * lhalf;
            #pragma unroll
            for (int ct = 0; ct < NT_; ++ct) {
                const __bf16* bp = WhT + (size_t)(c0 + ct * 16 + lmod) * U_ + kb;
                v16bf bm;
                ((v8bf*)&bm)[0] = *(const v8bf*)(bp);
                ((v8bf*)&bm)[1] = *(const v8bf*)(bp + 16);
                acc[ct] = __builtin_amdgcn_wmma_f32_16x16x32_bf16(
                    false, a, false, bm, (short)0, acc[ct], false, false);
            }
        }

        __syncthreads();   // all waves done reading old h / xs

        // ---- h_new = tanh(acc + b), store bf16 back to LDS ----
        // C layout: VGPR v, lanes0-15 => M=v ; lanes16-31 => M=8+v ; N = lane%16
        #pragma unroll
        for (int ct = 0; ct < NT_; ++ct) {
            #pragma unroll
            for (int v = 0; v < 8; ++v) {
                float val = TANHF_(acc[ct][v] + bias[ct]);
                hbuf[r0 + 8 * lhalf + v][c0 + ct * 16 + lmod] = (__bf16)val;
            }
        }
    }
    __syncthreads();

    // ---- out = h_last @ Wo + bo  (wave32 reduction) ----
    for (int r = wave * 4; r < wave * 4 + 4; ++r) {
        float s = 0.0f;
        for (int k = lane; k < U_; k += 32) s += (float)hbuf[r][k] * Wo[k];
        #pragma unroll
        for (int off = 16; off > 0; off >>= 1) s += __shfl_xor(s, off, 32);
        if (lane == 0) out[rowbase + r] = s + bo[0];
    }
}

// ---------- launch ----------

extern "C" void kernel_launch(void* const* d_in, const int* in_sizes, int n_in,
                              void* d_out, int out_size, void* d_ws, size_t ws_size,
                              hipStream_t stream) {
    const int*   inputs = (const int*)  d_in[0];
    const float* emb    = (const float*)d_in[1];
    const float* Wx     = (const float*)d_in[2];
    const float* Wh     = (const float*)d_in[3];
    const float* bg     = (const float*)d_in[4];
    const float* Wo     = (const float*)d_in[5];
    const float* bo     = (const float*)d_in[6];
    float* out = (float*)d_out;

    char* ws = (char*)d_ws;
    const size_t embp_bytes = (size_t)V_ * EP_ * 2;   // 2,560,000
    const size_t wxt_bytes  = (size_t)U_ * EP_ * 2;   //   262,144
    __bf16* embP = (__bf16*)(ws);
    __bf16* WxT  = (__bf16*)(ws + embp_bytes);
    __bf16* WhT  = (__bf16*)(ws + embp_bytes + wxt_bytes);  // +2,097,152 => ~4.9MB total

    prep_emb_k<<<(V_ * EP_ + 255) / 256, 256, 0, stream>>>(emb, embP);
    prep_wxt_k<<<(U_ * EP_ + 255) / 256, 256, 0, stream>>>(Wx, WxT);
    prep_wht_k<<<(U_ * U_  + 255) / 256, 256, 0, stream>>>(Wh, WhT);

    const size_t smem = (size_t)ROWS_ * U_ * 2 + (size_t)ROWS_ * EP_ * 2;  // 73728 B
    rnn_main_k<<<B_ / ROWS_, 256, smem, stream>>>(inputs, embP, WxT, WhT, bg, Wo, bo, out);
}
